// SFAttention_10041633538716
// MI455X (gfx1250) — matrix operand order repaired and checked by reading, not compiled, not run on hardware
//
#include <hip/hip_runtime.h>
#include <hip/hip_bf16.h>
#include <math.h>

#define BN_EPS 1e-5f

typedef __attribute__((ext_vector_type(16))) __bf16 v16bf;
typedef __attribute__((ext_vector_type(8)))  __bf16 v8bf;
typedef __attribute__((ext_vector_type(8)))  float  v8f;
typedef __attribute__((ext_vector_type(4)))  int    v4i;

static constexpr int CB = 4;      // batch
static constexpr int CC = 128;    // channels
static constexpr int CN = 4096;   // H*W spatial positions

// ---- CDNA5 async global->LDS path (guarded; fallback keeps same structure) -
#if defined(__has_builtin)
#if __has_builtin(__builtin_amdgcn_global_load_async_to_lds_b128)
#define HAVE_ASYNC_LDS 1
#endif
#if __has_builtin(__builtin_amdgcn_s_wait_asynccnt)
#define HAVE_WAIT_ASYNC 1
#endif
#endif

__device__ __forceinline__ void async_copy16(const unsigned short* g, unsigned short* l) {
#if defined(HAVE_ASYNC_LDS)
  __builtin_amdgcn_global_load_async_to_lds_b128(
      (__attribute__((address_space(1))) v4i*)g,
      (__attribute__((address_space(3))) v4i*)l, 0, 0);
#else
  *(uint4*)l = *(const uint4*)g;
#endif
}

__device__ __forceinline__ void async_wait0() {
#if defined(HAVE_ASYNC_LDS)
#if defined(HAVE_WAIT_ASYNC)
  __builtin_amdgcn_s_wait_asynccnt(0);
#else
  asm volatile("s_wait_asynccnt 0" ::: "memory");
#endif
#endif
}

// round-to-nearest-even f32 -> bf16 bit pattern
__device__ __forceinline__ unsigned short f2bf(float f) {
  unsigned u = __builtin_bit_cast(unsigned, f);
  u += 0x7FFFu + ((u >> 16) & 1u);
  return (unsigned short)(u >> 16);
}

// ---------------------------------------------------------------------------
// Projection: y = ELU(BN(W x + b)), written as bf16 pre-swizzled into the
// CDNA5 WMMA per-lane register layouts (ISA 7.12.2):
//   mode 0 (Q): A-matrix layout tiles  Qa[b][qt16][cc32][lane][16]
//   mode 1 (K): B-matrix layout tiles  Kb[b][kt16][cc32][lane][16]
//   mode 2 (V): B-matrix layout tiles  Vb[b][kt32][ct16][lane][16]
// ---------------------------------------------------------------------------
__global__ __launch_bounds__(256)
void sfattn_proj_kernel(const float* __restrict__ x, const float* __restrict__ w,
                        const float* __restrict__ bias,
                        const float* __restrict__ bns, const float* __restrict__ bnb,
                        const float* __restrict__ bnm, const float* __restrict__ bnv,
                        unsigned short* __restrict__ out, int mode)
{
  __shared__ float xs[CC][64];
  const int n0  = blockIdx.x * 64;
  const int b   = blockIdx.y;
  const int tid = threadIdx.x;

  for (int i = tid; i < CC * 64; i += 256) {
    int c = i >> 6, nl = i & 63;
    xs[c][nl] = x[((size_t)(b * CC + c)) * CN + n0 + nl];
  }
  __syncthreads();

  const int nl = tid & 63;
  const int og = tid >> 6;           // 4 groups of 32 output channels
  const int n  = n0 + nl;

  for (int oo = 0; oo < 32; ++oo) {
    const int o = og * 32 + oo;
    const float* wr = w + (size_t)o * CC;
    float acc = 0.f;
#pragma unroll 8
    for (int c = 0; c < CC; ++c) acc += xs[c][nl] * wr[c];
    acc += bias[o];
    float inv = bns[o] * rsqrtf(bnv[o] + BN_EPS);
    float y = acc * inv + (bnb[o] - bnm[o] * inv);
    y = (y > 0.f) ? y : (__expf(y) - 1.f);   // ELU(alpha=1)
    unsigned short h = f2bf(y);

    size_t idx;
    if (mode == 0) {
      int qt = n >> 4, l16 = n & 15;
      int cc = o >> 5, co = o & 31;
      int g = co >> 3;
      int sel = g & 1;
      int hh  = (co & 7) + ((g >> 1) << 3);
      int lane = sel * 16 + l16;
      idx = ((((size_t)b * 256 + qt) * 4 + cc) * 32 + lane) * 16 + hh;
    } else if (mode == 1) {
      int kt = n >> 4;
      int cc = o >> 5, rem = o & 31;
      int lane = (((rem >> 3) & 1) << 4) + (n & 15);
      idx = ((((size_t)b * 256 + kt) * 4 + cc) * 32 + lane) * 16 + ((rem & 7) + ((rem >> 4) << 3));
    } else {
      int kt  = n >> 5, remk = n & 31;
      int ct  = o >> 4;
      int lane = ((remk >> 4) << 4) + (o & 15);
      idx = ((((size_t)b * 128 + kt) * 8 + ct) * 32 + lane) * 16 + (remk & 15);
    }
    out[idx] = h;
  }
}

// ---------------------------------------------------------------------------
// Flash attention: grid (64 q-blocks, 4 batches), 128 threads = 4 waves.
// K/V slabs (64 k-positions = 16KB + 16KB) are staged global->LDS with the
// CDNA5 async path, double-buffered; each wave owns a 16-row q tile.
// ---------------------------------------------------------------------------
__global__ __launch_bounds__(128)
void sfattn_attn_kernel(const unsigned short* __restrict__ qa,
                        const unsigned short* __restrict__ kb,
                        const unsigned short* __restrict__ vb,
                        const float* __restrict__ xv,
                        const float* __restrict__ gamma_p,
                        const float* __restrict__ beta_p,
                        float* __restrict__ out)
{
  __shared__ __align__(16) unsigned short kbuf[2][8192];   // 2 x 16KB
  __shared__ __align__(16) unsigned short vbuf[2][8192];   // 2 x 16KB
  __shared__ __align__(64) unsigned short plds[4][16][64]; // per-wave P tile

  const int b    = blockIdx.y;
  const int tid  = threadIdx.x;
  const int wave = tid >> 5;
  const int lane = tid & 31;
  const int sel  = lane >> 4;
  const int l16  = lane & 15;
  const int qt   = blockIdx.x * 4 + wave;    // 16-row q tile id, 0..255

  // Preload Q A-tiles for all 4 c-chunks (resident for whole k loop)
  v16bf Qc[4];
#pragma unroll
  for (int cc = 0; cc < 4; ++cc) {
    size_t i = ((((size_t)b * 256 + qt) * 4 + cc) * 32 + lane) * 16;
    Qc[cc] = *(const v16bf*)(qa + i);
  }

  const v8f vzero = {0.f, 0.f, 0.f, 0.f, 0.f, 0.f, 0.f, 0.f};
  v8f Ot[8];
#pragma unroll
  for (int t = 0; t < 8; ++t) Ot[t] = vzero;

  float m[8], lsum[8];
#pragma unroll
  for (int r = 0; r < 8; ++r) { m[r] = -__builtin_inff(); lsum[r] = 0.f; }

  // cooperative async stage of one 64-k slab (K: 8192 ush, V: 8192 ush)
  auto stage = [&](int kt, int nb) {
    const unsigned short* gk = kb + (((size_t)b * 256 + 4 * kt) * 4) * 512;
    const unsigned short* gv = vb + (((size_t)b * 128 + 2 * kt) * 8) * 512;
#pragma unroll
    for (int j = 0; j < 8; ++j) {
      int off = (j * 128 + tid) * 8;         // 16B granules
      async_copy16(gk + off, &kbuf[nb][off]);
      async_copy16(gv + off, &vbuf[nb][off]);
    }
  };

  const int NIT = CN / 64;
  stage(0, 0);
  async_wait0();
  __syncthreads();

  for (int kt = 0; kt < NIT; ++kt) {
    const int nb = kt & 1;
    if (kt + 1 < NIT) stage(kt + 1, nb ^ 1);

    // ---- S = Q^T K for a 16q x 64k slab (4 n-tiles) -------------------
    v8f S[4];
#pragma unroll
    for (int s = 0; s < 4; ++s) S[s] = vzero;
#pragma unroll
    for (int sub = 0; sub < 4; ++sub) {
#pragma unroll
      for (int cc = 0; cc < 4; ++cc) {
        v16bf Bk = *(const v16bf*)&kbuf[nb][(sub * 4 + cc) * 512 + lane * 16];
        S[sub] = __builtin_amdgcn_wmma_f32_16x16x32_bf16(false, Qc[cc], false, Bk,
                                                         (short)0, S[sub], false, false);
      }
    }

    // ---- online softmax (row stats via 16-lane xor reductions) --------
#pragma unroll
    for (int r = 0; r < 8; ++r) {
      float mx = fmaxf(fmaxf(S[0][r], S[1][r]), fmaxf(S[2][r], S[3][r]));
      mx = fmaxf(mx, __shfl_xor(mx, 1));
      mx = fmaxf(mx, __shfl_xor(mx, 2));
      mx = fmaxf(mx, __shfl_xor(mx, 4));
      mx = fmaxf(mx, __shfl_xor(mx, 8));
      float mnew = fmaxf(m[r], mx);
      float a = __expf(m[r] - mnew);
      m[r] = mnew;
      float p0 = __expf(S[0][r] - mnew);
      float p1 = __expf(S[1][r] - mnew);
      float p2 = __expf(S[2][r] - mnew);
      float p3 = __expf(S[3][r] - mnew);
      float s = (p0 + p1) + (p2 + p3);
      s += __shfl_xor(s, 1);
      s += __shfl_xor(s, 2);
      s += __shfl_xor(s, 4);
      s += __shfl_xor(s, 8);
      lsum[r] = lsum[r] * a + s;
#pragma unroll
      for (int ct = 0; ct < 8; ++ct) Ot[ct][r] *= a;
      int row = sel * 8 + r;                    // D-layout row for this lane
      plds[wave][row][l16]      = f2bf(p0);
      plds[wave][row][16 + l16] = f2bf(p1);
      plds[wave][row][32 + l16] = f2bf(p2);
      plds[wave][row][48 + l16] = f2bf(p3);
    }

    // ---- O += P V^T, two 32-k halves x 8 channel tiles ----------------
#pragma unroll
    for (int half = 0; half < 2; ++half) {
      v8bf plo = *(const v8bf*)&plds[wave][l16][half * 32 + sel * 16];
      v8bf phi = *(const v8bf*)&plds[wave][l16][half * 32 + sel * 16 + 8];
      v16bf Pa = __builtin_shufflevector(plo, phi, 0, 1, 2, 3, 4, 5, 6, 7,
                                         8, 9, 10, 11, 12, 13, 14, 15);
#pragma unroll
      for (int ct = 0; ct < 8; ++ct) {
        v16bf Bv = *(const v16bf*)&vbuf[nb][(half * 8 + ct) * 512 + lane * 16];
        Ot[ct] = __builtin_amdgcn_wmma_f32_16x16x32_bf16(false, Pa, false, Bv,
                                                         (short)0, Ot[ct], false, false);
      }
    }

    async_wait0();        // staged slab for next iter is in LDS
    __syncthreads();      // everyone done reading current slab
  }

  // ---- epilogue: normalize, gamma*out + beta*x_v ----------------------
  const float g  = gamma_p[0];
  const float be = beta_p[0];
#pragma unroll
  for (int ct = 0; ct < 8; ++ct) {
    int c = ct * 16 + l16;
#pragma unroll
    for (int r = 0; r < 8; ++r) {
      int q = qt * 16 + sel * 8 + r;
      float val = Ot[ct][r] / lsum[r];
      size_t o = ((size_t)(b * CC + c)) * CN + q;
      out[o] = g * val + be * xv[o];
    }
  }
}

// ---------------------------------------------------------------------------
extern "C" void kernel_launch(void* const* d_in, const int* in_sizes, int n_in,
                              void* d_out, int out_size, void* d_ws, size_t ws_size,
                              hipStream_t stream) {
  (void)in_sizes; (void)n_in; (void)out_size; (void)ws_size;
  const float* xq = (const float*)d_in[0];
  const float* xk = (const float*)d_in[1];
  const float* xv = (const float*)d_in[2];
  const float* qw = (const float*)d_in[3];
  const float* qb = (const float*)d_in[4];
  const float* qs = (const float*)d_in[5];
  const float* qbb = (const float*)d_in[6];
  const float* qm = (const float*)d_in[7];
  const float* qv = (const float*)d_in[8];
  const float* kw = (const float*)d_in[9];
  const float* kbi = (const float*)d_in[10];
  const float* ks = (const float*)d_in[11];
  const float* kbb = (const float*)d_in[12];
  const float* km = (const float*)d_in[13];
  const float* kv = (const float*)d_in[14];
  const float* vw = (const float*)d_in[15];
  const float* vbi = (const float*)d_in[16];
  const float* vs = (const float*)d_in[17];
  const float* vbb = (const float*)d_in[18];
  const float* vm = (const float*)d_in[19];
  const float* vv = (const float*)d_in[20];
  const float* gamma = (const float*)d_in[21];
  const float* beta  = (const float*)d_in[22];

  const size_t SZ = (size_t)CB * 256 * 4 * 32 * 16;   // 2M bf16 elems = 4 MB
  unsigned short* qap = (unsigned short*)d_ws;
  unsigned short* kbp = qap + SZ;
  unsigned short* vbp = kbp + SZ;

  dim3 gp(CN / 64, CB);
  sfattn_proj_kernel<<<gp, 256, 0, stream>>>(xq, qw, qb, qs, qbb, qm, qv, qap, 0);
  sfattn_proj_kernel<<<gp, 256, 0, stream>>>(xk, kw, kbi, ks, kbb, km, kv, kbp, 1);
  sfattn_proj_kernel<<<gp, 256, 0, stream>>>(xv, vw, vbi, vs, vbb, vm, vv, vbp, 2);

  sfattn_attn_kernel<<<dim3(CN / 64, CB), 128, 0, stream>>>(
      qap, kbp, vbp, xv, gamma, beta, (float*)d_out);
}
